// GenuineTrainedDistillationTransformer_12592844112101
// MI455X (gfx1250) — matrix operand order repaired and checked by reading, not compiled, not run on hardware
//
#include <hip/hip_runtime.h>
#include <hip/hip_bf16.h>

// ---------------------------------------------------------------------------
// CDNA5 (gfx1250) full-model forward:
//   embed -> leech enc/dec -> 3x [LN -> QKV (WMMA bf16) -> FFT filter ->
//   flash attention (WMMA) -> O proj (WMMA) -> norm-residual -> LN ->
//   FFN (WMMA, gelu) -> residual] -> mean-pool -> classifier
// Heavy GEMMs: v_wmma_f32_16x16x32_bf16, LDS-staged; the GEMM mainloop uses
// GLOBAL_LOAD_ASYNC_TO_LDS_B128 with a 2-deep pipeline synchronized by
// s_wait_asynccnt (ASYNCcnt, ISA ch.08 §4) when the builtin is available.
// ---------------------------------------------------------------------------

typedef __attribute__((ext_vector_type(16))) __bf16 v16bf;
typedef __attribute__((ext_vector_type(8)))  float  v8f;
typedef int v4i_gcc __attribute__((vector_size(16)));  // matches builtin proto

#if defined(__HIP_DEVICE_COMPILE__) &&                                   \
    __has_builtin(__builtin_amdgcn_global_load_async_to_lds_b128) &&     \
    __has_builtin(__builtin_amdgcn_s_wait_asynccnt)
#define USE_ASYNC_LDS 1
#else
#define USE_ASYNC_LDS 0
#endif

__device__ __forceinline__ v8f wmma_bf16(v16bf a, v16bf b, v8f c) {
  return __builtin_amdgcn_wmma_f32_16x16x32_bf16(false, a, false, b,
                                                 (short)0, c, false, false);
}

// A-fragment (16x32 bf16): lane m = l%16; elems 0..7 = K base+0..7,
// elems 8..15 = K base+16..23, base = (l>=16 ? 8 : 0) [+ks*32].
__device__ __forceinline__ v16bf load_a_frag(const __bf16* p) {
  union { float4 f4[2]; v16bf v; } u;
  u.f4[0] = *reinterpret_cast<const float4*>(p);
  u.f4[1] = *reinterpret_cast<const float4*>(p + 16);
  return u.v;
}
// B-fragment (32x16 bf16) from [n][k]-layout LDS: lane n = l%16,
// elems 0..7 = K base+0..7, elems 8..15 = K base+16..23, base = (l>=16 ? 8 : 0).
__device__ __forceinline__ v16bf load_b_frag(const __bf16* p) {
  union { float4 f4[2]; v16bf v; } u;
  u.f4[0] = *reinterpret_cast<const float4*>(p);
  u.f4[1] = *reinterpret_cast<const float4*>(p + 16);
  return u.v;
}

#if USE_ASYNC_LDS
// 16-byte async global->LDS copy (no VGPR transit; tracked by ASYNCcnt).
// Builtin proto (from clang diagnostic): param1 = v4i32* in global AS.
__device__ __forceinline__ void async_copy16(const __bf16* g, __bf16* l) {
  __builtin_amdgcn_global_load_async_to_lds_b128(
      (__attribute__((address_space(1))) v4i_gcc*)(g),
      (__attribute__((address_space(3))) v4i_gcc*)(l), 0, 0);
}
#endif

__device__ __forceinline__ float gelu_exact(float x) {
  return 0.5f * x * (1.0f + erff(x * 0.70710678118654752f));
}

__device__ __forceinline__ float block_reduce_sum_256(float v, float* red) {
  int tid = threadIdx.x;
  red[tid] = v;
  __syncthreads();
  for (int s = 128; s > 0; s >>= 1) {
    if (tid < s) red[tid] += red[tid + s];
    __syncthreads();
  }
  float r = red[0];
  __syncthreads();
  return r;
}

// ---------------------------------------------------------------------------
// Tiled GEMM: C[M,N] = A[M,K](bf16, row-major, ldA) @ Bt[N,K](bf16) + bias[N]
// EPI 0: write fp32. EPI 1: gelu then write bf16 (for FFN1 -> FFN2 chaining).
// Block 256 thr = 8 waves, 64x64 tile, K-step 32; each wave: 1 m-tile x 2
// n-tiles. Mainloop double-buffers LDS stages via async copies when available.
// ---------------------------------------------------------------------------
template <int EPI>
__global__ __launch_bounds__(256) void gemm_bf16_kernel(
    const __bf16* __restrict__ A, const __bf16* __restrict__ Bt,
    const float* __restrict__ bias, float* __restrict__ Cf,
    __bf16* __restrict__ Cb, int M, int N, int K, int ldA) {
  __shared__ alignas(16) __bf16 sA[2][64 * 32];
  __shared__ alignas(16) __bf16 sB[2][64 * 32];
  const int tid = threadIdx.x;
  const int w = tid >> 5;
  const int lane = tid & 31;
  const int n0 = blockIdx.x * 64;
  const int m0 = blockIdx.y * 64;
  const int mt = w >> 1;       // 0..3
  const int ng = w & 1;        // 0..1 -> 32 cols
  const int c0a = (lane & 16) ? 8 : 0;
  const int c0b = (lane & 16) ? 8 : 0;
  const int am = mt * 16 + (lane & 15);
  const int bn0 = (ng * 2) * 16 + (lane & 15);
  const int bn1 = (ng * 2 + 1) * 16 + (lane & 15);
  v8f acc0 = {}, acc1 = {};

  const int e = tid * 8;       // 8 bf16 = 16B per thread per tile
  const int er = e >> 5, ec = e & 31;
  const __bf16* gA = A + (size_t)(m0 + er) * ldA + ec;
  const __bf16* gB = Bt + (size_t)(n0 + er) * K + ec;
  const int nsteps = K / 32;

#if USE_ASYNC_LDS
  // 2-deep async pipeline: stage i+1 in flight while stage i computes.
  async_copy16(gA, &sA[0][e]);
  async_copy16(gB, &sB[0][e]);
  for (int i = 0; i < nsteps; ++i) {
    if (i + 1 < nsteps) {
      async_copy16(gA + (i + 1) * 32, &sA[(i + 1) & 1][e]);
      async_copy16(gB + (i + 1) * 32, &sB[(i + 1) & 1][e]);
      __builtin_amdgcn_s_wait_asynccnt(2);  // stage i landed (in-order)
    } else {
      __builtin_amdgcn_s_wait_asynccnt(0);
    }
    __syncthreads();
    const __bf16* bufA = sA[i & 1];
    const __bf16* bufB = sB[i & 1];
    v16bf af = load_a_frag(&bufA[am * 32 + c0a]);
    v16bf b0 = load_b_frag(&bufB[bn0 * 32 + c0b]);
    v16bf b1 = load_b_frag(&bufB[bn1 * 32 + c0b]);
    acc0 = wmma_bf16(af, b0, acc0);
    acc1 = wmma_bf16(af, b1, acc1);
    __syncthreads();  // WAR guard before next stage writes this buffer
  }
#else
  for (int i = 0; i < nsteps; ++i) {
    *reinterpret_cast<float4*>(&sA[0][e]) =
        *reinterpret_cast<const float4*>(gA + i * 32);
    *reinterpret_cast<float4*>(&sB[0][e]) =
        *reinterpret_cast<const float4*>(gB + i * 32);
    __syncthreads();
    v16bf af = load_a_frag(&sA[0][am * 32 + c0a]);
    v16bf b0 = load_b_frag(&sB[0][bn0 * 32 + c0b]);
    v16bf b1 = load_b_frag(&sB[0][bn1 * 32 + c0b]);
    acc0 = wmma_bf16(af, b0, acc0);
    acc1 = wmma_bf16(af, b1, acc1);
    __syncthreads();
  }
#endif

  const int hi8 = (lane & 16) ? 8 : 0;
  const int ncol0 = n0 + ng * 32 + (lane & 15);
  const int ncol1 = ncol0 + 16;
  const float bb0 = bias ? bias[ncol0] : 0.0f;
  const float bb1 = bias ? bias[ncol1] : 0.0f;
#pragma unroll
  for (int r = 0; r < 8; ++r) {
    const size_t mrow = (size_t)(m0 + mt * 16 + r + hi8);
    float v0 = acc0[r] + bb0;
    float v1 = acc1[r] + bb1;
    if (EPI == 1) {
      Cb[mrow * N + ncol0] = (__bf16)gelu_exact(v0);
      Cb[mrow * N + ncol1] = (__bf16)gelu_exact(v1);
    } else {
      Cf[mrow * N + ncol0] = v0;
      Cf[mrow * N + ncol1] = v1;
    }
  }
}

// ---------------------------------------------------------------------------
// Flash attention per (b,h): 64-query block vs 64-key blocks, online softmax.
// q/k/v layout: [B,T,H,Dh] fp32 (already spectrally filtered).
// S = Q K^T and O += P V both via WMMA bf16.
// ---------------------------------------------------------------------------
__global__ __launch_bounds__(256) void flash_attn_kernel(
    const float* __restrict__ q, const float* __restrict__ k,
    const float* __restrict__ v, float* __restrict__ att, int Bn, int Tn,
    int Hn, int Dhn) {
  __shared__ alignas(16) __bf16 sQ[64 * 128];
  __shared__ alignas(16) __bf16 sK[64 * 128];   // [key][d]  (== B^T [n][k])
  __shared__ alignas(16) __bf16 sVt[128 * 64];  // [d][key]  (B [n][k] for PV)
  __shared__ alignas(16) __bf16 sSb[64 * 64];   // S then P (bf16)
  __shared__ float sM[64], sL[64], sC[64];

  const int tid = threadIdx.x;
  const int w = tid >> 5;
  const int lane = tid & 31;
  const int bh = blockIdx.y;
  const int bidx = bh / Hn;
  const int hh = bh % Hn;
  const int t0 = blockIdx.x * 64;
  const size_t strT = (size_t)Hn * Dhn;  // 1024
  const float* qp = q + (size_t)bidx * Tn * strT + (size_t)hh * Dhn;
  const float* kp = k + (size_t)bidx * Tn * strT + (size_t)hh * Dhn;
  const float* vp = v + (size_t)bidx * Tn * strT + (size_t)hh * Dhn;

  for (int idx = tid; idx < 64 * 128; idx += 256) {
    const int m = idx >> 7, d = idx & 127;
    sQ[idx] = (__bf16)qp[(size_t)(t0 + m) * strT + d];
  }
  if (tid < 64) { sM[tid] = -1e30f; sL[tid] = 0.0f; }

  const int mt = w >> 1;   // m-tile 0..3
  const int ng = w & 1;    // O covers d-cols [ng*64, ng*64+64)
  const int nt0 = ng * 2;  // S n-tiles {nt0, nt0+1}
  const int c0a = (lane & 16) ? 8 : 0;
  const int c0b = (lane & 16) ? 8 : 0;
  const int hi8 = (lane & 16) ? 8 : 0;
  const int mrow = mt * 16 + (lane & 15);
  const float scale = 0.088388347648318447f;  // 1/sqrt(128)
  v8f o0 = {}, o1 = {}, o2 = {}, o3 = {};
  __syncthreads();

  for (int kb = 0; kb < Tn / 64; ++kb) {
    for (int idx = tid; idx < 64 * 128; idx += 256) {
      const int j = idx >> 7, d = idx & 127;
      sK[idx] = (__bf16)kp[(size_t)(kb * 64 + j) * strT + d];
      sVt[d * 64 + j] = (__bf16)vp[(size_t)(kb * 64 + j) * strT + d];
    }
    __syncthreads();

    // S tile: Q[64x128] @ K^T -> [64x64]
    v8f s0 = {}, s1 = {};
#pragma unroll
    for (int ks = 0; ks < 4; ++ks) {
      v16bf af = load_a_frag(&sQ[mrow * 128 + ks * 32 + c0a]);
      v16bf b0 = load_b_frag(&sK[(nt0 * 16 + (lane & 15)) * 128 + ks * 32 + c0b]);
      v16bf b1 = load_b_frag(&sK[((nt0 + 1) * 16 + (lane & 15)) * 128 + ks * 32 + c0b]);
      s0 = wmma_bf16(af, b0, s0);
      s1 = wmma_bf16(af, b1, s1);
    }
#pragma unroll
    for (int r = 0; r < 8; ++r) {
      const int mm = mt * 16 + r + hi8;
      sSb[mm * 64 + nt0 * 16 + (lane & 15)] = (__bf16)(s0[r] * scale);
      sSb[mm * 64 + (nt0 + 1) * 16 + (lane & 15)] = (__bf16)(s1[r] * scale);
    }
    __syncthreads();

    // Online softmax row update (one thread per query row).
    if (tid < 64) {
      float mo = sM[tid];
      float mx = mo;
      for (int j = 0; j < 64; ++j) mx = fmaxf(mx, (float)sSb[tid * 64 + j]);
      const float corr = __expf(mo - mx);
      float ps = 0.0f;
      for (int j = 0; j < 64; ++j) {
        const float p = __expf((float)sSb[tid * 64 + j] - mx);
        sSb[tid * 64 + j] = (__bf16)p;
        ps += p;
      }
      sM[tid] = mx;
      sL[tid] = sL[tid] * corr + ps;
      sC[tid] = corr;
    }
    __syncthreads();

    // Rescale O, then O += P @ V.
    float cr[8];
#pragma unroll
    for (int r = 0; r < 8; ++r) cr[r] = sC[mt * 16 + r + hi8];
#pragma unroll
    for (int r = 0; r < 8; ++r) {
      o0[r] *= cr[r]; o1[r] *= cr[r]; o2[r] *= cr[r]; o3[r] *= cr[r];
    }
#pragma unroll
    for (int ks = 0; ks < 2; ++ks) {
      v16bf af = load_a_frag(&sSb[mrow * 64 + ks * 32 + c0a]);
      v16bf b0 = load_b_frag(&sVt[(ng * 64 + 0 * 16 + (lane & 15)) * 64 + ks * 32 + c0b]);
      v16bf b1 = load_b_frag(&sVt[(ng * 64 + 1 * 16 + (lane & 15)) * 64 + ks * 32 + c0b]);
      v16bf b2 = load_b_frag(&sVt[(ng * 64 + 2 * 16 + (lane & 15)) * 64 + ks * 32 + c0b]);
      v16bf b3 = load_b_frag(&sVt[(ng * 64 + 3 * 16 + (lane & 15)) * 64 + ks * 32 + c0b]);
      o0 = wmma_bf16(af, b0, o0);
      o1 = wmma_bf16(af, b1, o1);
      o2 = wmma_bf16(af, b2, o2);
      o3 = wmma_bf16(af, b3, o3);
    }
    __syncthreads();
  }

  float lr[8];
#pragma unroll
  for (int r = 0; r < 8; ++r) lr[r] = 1.0f / sL[mt * 16 + r + hi8];
#pragma unroll
  for (int r = 0; r < 8; ++r) {
    const size_t row = (size_t)(bidx * Tn + t0 + mt * 16 + r + hi8);
    float* op = att + (row * Hn + hh) * Dhn + ng * 64 + (lane & 15);
    op[0] = o0[r] * lr[r];
    op[16] = o1[r] * lr[r];
    op[32] = o2[r] * lr[r];
    op[48] = o3[r] * lr[r];
  }
}

// ---------------------------------------------------------------------------
// Radix-2 DIT FFT over T=1024 per (b,h,d) line, multiply by real spectral
// filter exp(-aa*arctan(log|f|)), keep real part, write back in place.
// ---------------------------------------------------------------------------
__global__ __launch_bounds__(256) void fft_filter_kernel(
    float* __restrict__ data, const float* __restrict__ alpha,
    const float* __restrict__ fscale, int layer, int Bn, int Tn, int Hn,
    int Dhn) {
  __shared__ float re[1024];
  __shared__ float im[1024];
  const int line = blockIdx.x;
  const int d = line % Dhn;
  const int hh = (line / Dhn) % Hn;
  const int bidx = line / (Dhn * Hn);
  const size_t str = (size_t)Hn * Dhn;
  const size_t base = (size_t)bidx * Tn * str + (size_t)hh * Dhn + d;
  const int tid = threadIdx.x;

  for (int t = tid; t < 1024; t += 256) {
    const int rv = (int)(__brev((unsigned)t) >> 22);
    re[t] = data[base + (size_t)rv * str];
    im[t] = 0.0f;
  }
  __syncthreads();

  for (int len = 2; len <= 1024; len <<= 1) {
    const int half = len >> 1;
    for (int i = tid; i < 512; i += 256) {
      const int blk = i / half, pos = i % half;
      const int i1 = blk * len + pos, i2 = i1 + half;
      const float ang = -6.28318530717958647f * (float)pos / (float)len;
      const float c = __cosf(ang), sn = __sinf(ang);
      const float xr = re[i2], xi = im[i2];
      const float tr = c * xr - sn * xi;
      const float ti = c * xi + sn * xr;
      re[i2] = re[i1] - tr;
      im[i2] = im[i1] - ti;
      re[i1] += tr;
      im[i1] += ti;
    }
    __syncthreads();
  }

  const float aa = alpha[layer] + fscale[layer] * (1.5f - 1.5f);  // FRACTAL_DIM
  for (int t = tid; t < 1024; t += 256) {
    const float f = (t < 512) ? (float)t / 1024.0f : ((float)t - 1024.0f) / 1024.0f;
    const float km = fmaxf(fabsf(f), 1e-10f);
    const float filt = __expf(-aa * atanf(__logf(km)));
    data[base + (size_t)t * str] = re[t] * filt;
  }
}

// ---------------------------------------------------------------------------
// Elementwise / row-wise helpers
// ---------------------------------------------------------------------------
__global__ __launch_bounds__(256) void embed_kernel(
    const int* __restrict__ ids, const float* __restrict__ tok_table,
    const float* __restrict__ emb_scales, const float* __restrict__ emb_norm,
    const float* __restrict__ pos_emb, float* __restrict__ x, int Tn, int D) {
  __shared__ float red[256];
  const size_t row = blockIdx.x;
  const int t = (int)(row % Tn);
  const int id = ids[row];
  const float* tk = tok_table + (size_t)id * D;
  float tv[4];
  float st = 0.0f, se = 0.0f;
#pragma unroll
  for (int i = 0; i < 4; ++i) {
    const int d = threadIdx.x + i * 256;
    const float tvv = tk[d];
    tv[i] = tvv;
    st += tvv * tvv;
    const float e = tvv * emb_scales[d];
    se += e * e;
  }
  const float nt = sqrtf(block_reduce_sum_256(st, red));
  const float ne = sqrtf(block_reduce_sum_256(se, red));
  const float sc = (nt / (ne + 1e-8f)) * emb_norm[0];
#pragma unroll
  for (int i = 0; i < 4; ++i) {
    const int d = threadIdx.x + i * 256;
    x[row * D + d] = tv[i] * emb_scales[d] * sc + pos_emb[(size_t)t * D + d];
  }
}

__global__ __launch_bounds__(256) void leech_encode_kernel(
    const float* __restrict__ x, const float* __restrict__ W_el,
    const float* __restrict__ b_el, const float* __restrict__ ecs_p,
    const float* __restrict__ epres_p, float* __restrict__ enc, int D) {
  __shared__ float red[256];
  __shared__ float lp[24];
  __shared__ float s_nlp;
  const size_t row = blockIdx.x;
  const float* xr = x + row * D;
  float ss = 0.0f;
  for (int d = threadIdx.x; d < D; d += 256) {
    const float vv = xr[d];
    ss += vv * vv;
  }
  const float ie = sqrtf(block_reduce_sum_256(ss, red));
  const float ecs = ecs_p[0], epres = epres_p[0];
  if (threadIdx.x < 24) {
    float p = b_el[threadIdx.x];
    for (int d = 0; d < D; ++d) p += xr[d] * W_el[d * 24 + threadIdx.x];
    lp[threadIdx.x] = rintf(p / ecs) * ecs;
  }
  __syncthreads();
  if (threadIdx.x == 0) {
    float n = 0.0f;
    for (int j = 0; j < 24; ++j) n += lp[j] * lp[j];
    s_nlp = sqrtf(n);
  }
  __syncthreads();
  if (threadIdx.x < 24)
    enc[row * 24 + threadIdx.x] = lp[threadIdx.x] * (ie / (s_nlp + 1e-8f)) * epres;
}

__global__ __launch_bounds__(256) void leech_decode_kernel(
    float* __restrict__ x, const float* __restrict__ enc,
    const float* __restrict__ W_le, const float* __restrict__ b_le,
    const float* __restrict__ ecs_p, const float* __restrict__ epres_p,
    int D) {
  __shared__ float red[256];
  __shared__ float corr[24];
  __shared__ float dec[1024];
  __shared__ float s_ie2;
  const size_t row = blockIdx.x;
  const float ecs = ecs_p[0], epres = epres_p[0];
  if (threadIdx.x < 24) {
    const float e = enc[row * 24 + threadIdx.x];
    corr[threadIdx.x] = (fabsf(e) > ecs) ? e : 0.0f;
  }
  __syncthreads();
  if (threadIdx.x == 0) {
    float n = 0.0f;
    for (int j = 0; j < 24; ++j) {
      const float e = enc[row * 24 + j];
      n += e * e;
    }
    s_ie2 = sqrtf(n);
  }
  __syncthreads();
  float ssd = 0.0f;
  for (int d = threadIdx.x; d < D; d += 256) {
    float acc = b_le[d];
    for (int j = 0; j < 24; ++j) acc += corr[j] * W_le[j * D + d];
    dec[d] = acc;
    ssd += acc * acc;
  }
  const float nd = sqrtf(block_reduce_sum_256(ssd, red));
  const float sc = (s_ie2 / (nd + 1e-8f)) * epres;
  for (int d = threadIdx.x; d < D; d += 256) x[row * D + d] += dec[d] * sc;
}

__global__ __launch_bounds__(256) void layernorm_kernel(
    const float* __restrict__ x, const float* __restrict__ w,
    const float* __restrict__ b, float* __restrict__ h,
    __bf16* __restrict__ hb, int D) {
  __shared__ float red[256];
  const size_t row = blockIdx.x;
  const float* xr = x + row * D;
  float loc[4];
  float s = 0.0f;
#pragma unroll
  for (int i = 0; i < 4; ++i) {
    loc[i] = xr[threadIdx.x + i * 256];
    s += loc[i];
  }
  const float mean = block_reduce_sum_256(s, red) * (1.0f / 1024.0f);
  float vs = 0.0f;
#pragma unroll
  for (int i = 0; i < 4; ++i) {
    const float dl = loc[i] - mean;
    vs += dl * dl;
  }
  const float var = block_reduce_sum_256(vs, red) * (1.0f / 1024.0f);
  const float inv = rsqrtf(var + 1e-5f);
#pragma unroll
  for (int i = 0; i < 4; ++i) {
    const int d = threadIdx.x + i * 256;
    const float y = (loc[i] - mean) * inv * w[d] + b[d];
    h[row * D + d] = y;
    hb[row * D + d] = (__bf16)y;
  }
}

__global__ __launch_bounds__(256) void resnorm_add_kernel(
    float* __restrict__ x, const float* __restrict__ h,
    const float* __restrict__ out, const float* __restrict__ enorm, int layer,
    int D) {
  __shared__ float red[256];
  const size_t row = blockIdx.x;
  float ov[4];
  float sh = 0.0f, so = 0.0f;
#pragma unroll
  for (int i = 0; i < 4; ++i) {
    const int d = threadIdx.x + i * 256;
    const float hv = h[row * D + d];
    sh += hv * hv;
    const float o = out[row * D + d];
    ov[i] = o;
    so += o * o;
  }
  const float nh = sqrtf(block_reduce_sum_256(sh, red));
  const float no = sqrtf(block_reduce_sum_256(so, red));
  const float sc = (nh / (no + 1e-8f)) * enorm[layer];
#pragma unroll
  for (int i = 0; i < 4; ++i) {
    const int d = threadIdx.x + i * 256;
    x[row * D + d] += ov[i] * sc;
  }
}

__global__ __launch_bounds__(256) void add_kernel(float* __restrict__ x,
                                                  const float* __restrict__ y,
                                                  size_t n) {
  const size_t i = (size_t)blockIdx.x * 256 + threadIdx.x;
  if (i < n) x[i] += y[i];
}

__global__ __launch_bounds__(256) void f32_to_bf16_kernel(
    const float* __restrict__ src, __bf16* __restrict__ dst, size_t n) {
  const size_t i = (size_t)blockIdx.x * 256 + threadIdx.x;
  if (i < n) dst[i] = (__bf16)src[i];
}

// Wt[n*K + k] = W[k*N + n]  (fp32 -> bf16, transposed for WMMA B [n][k])
__global__ __launch_bounds__(256) void convert_transpose_kernel(
    const float* __restrict__ W, __bf16* __restrict__ Wt, int K, int N) {
  const size_t idx = (size_t)blockIdx.x * 256 + threadIdx.x;
  if (idx >= (size_t)K * N) return;
  const size_t n = idx / K;
  const size_t k = idx % K;
  // Strided gather: prefetch a few rows ahead (speculative, drop-on-fault).
  __builtin_prefetch(W + (k + 8) * N + n, 0, 0);
  Wt[n * K + k] = (__bf16)W[k * N + n];
}

__global__ __launch_bounds__(256) void pool_classify_kernel(
    const float* __restrict__ x, const float* __restrict__ clsW,
    const float* __restrict__ clsb, float* __restrict__ out, int Tn, int D,
    int NCn) {
  __shared__ float red[256];
  __shared__ float pooled[1024];
  const int b = blockIdx.x;
  for (int d = threadIdx.x; d < D; d += 256) {
    float s = 0.0f;
    for (int t = 0; t < Tn; ++t) s += x[((size_t)b * Tn + t) * D + d];
    pooled[d] = s / (float)Tn;
  }
  __syncthreads();
  for (int c = 0; c < NCn; ++c) {
    float p = 0.0f;
    for (int d = threadIdx.x; d < D; d += 256) p += pooled[d] * clsW[d * NCn + c];
    const float tot = block_reduce_sum_256(p, red);
    if (threadIdx.x == 0) out[b * NCn + c] = tot + clsb[c];
  }
}

// ---------------------------------------------------------------------------
// Host orchestration
// ---------------------------------------------------------------------------
extern "C" void kernel_launch(void* const* d_in, const int* in_sizes, int n_in,
                              void* d_out, int out_size, void* d_ws,
                              size_t ws_size, hipStream_t stream) {
  (void)in_sizes; (void)n_in; (void)out_size; (void)ws_size;
  const int B = 4, T = 1024, D = 1024, H = 8, Dh = 128, L = 3, LAT = 24,
            NC = 2, D4 = 4096;
  const int M = B * T;  // 4096 rows

  const int*   input_ids  = (const int*)d_in[0];
  const float* tok_table  = (const float*)d_in[1];
  const float* emb_scales = (const float*)d_in[2];
  const float* emb_norm   = (const float*)d_in[3];
  const float* pos_emb    = (const float*)d_in[4];
  const float* W_el       = (const float*)d_in[5];
  const float* b_el       = (const float*)d_in[6];
  const float* W_le       = (const float*)d_in[7];
  const float* b_le       = (const float*)d_in[8];
  const float* ecs        = (const float*)d_in[9];
  const float* epres      = (const float*)d_in[10];
  const float* ln1_w      = (const float*)d_in[11];
  const float* ln1_b      = (const float*)d_in[12];
  const float* ln2_w      = (const float*)d_in[13];
  const float* ln2_b      = (const float*)d_in[14];
  const float* qW         = (const float*)d_in[15];
  const float* qb         = (const float*)d_in[16];
  const float* kW         = (const float*)d_in[17];
  const float* kb         = (const float*)d_in[18];
  const float* vW         = (const float*)d_in[19];
  const float* vb         = (const float*)d_in[20];
  const float* oW         = (const float*)d_in[21];
  const float* ob         = (const float*)d_in[22];
  const float* f1W        = (const float*)d_in[23];
  const float* f1b        = (const float*)d_in[24];
  const float* f2W        = (const float*)d_in[25];
  const float* f2b        = (const float*)d_in[26];
  const float* alpha      = (const float*)d_in[27];
  const float* fscale     = (const float*)d_in[28];
  const float* enorm      = (const float*)d_in[29];
  const float* clsW       = (const float*)d_in[30];
  const float* clsb       = (const float*)d_in[31];

  char* ws = (char*)d_ws;
  float*  x    = (float*)ws;  ws += (size_t)M * D * 4;    // 16 MB
  float*  h    = (float*)ws;  ws += (size_t)M * D * 4;    // 16 MB
  float*  qbuf = (float*)ws;  ws += (size_t)M * D * 4;    // 16 MB (also "out")
  float*  kbuf = (float*)ws;  ws += (size_t)M * D * 4;    // 16 MB
  float*  vbuf = (float*)ws;  ws += (size_t)M * D * 4;    // 16 MB
  float*  att  = (float*)ws;  ws += (size_t)M * D * 4;    // 16 MB
  __bf16* hb   = (__bf16*)ws; ws += (size_t)M * D * 2;    //  8 MB
  __bf16* actb = (__bf16*)ws; ws += (size_t)M * D4 * 2;   // 32 MB (attb/midb)
  __bf16* wbT  = (__bf16*)ws; ws += (size_t)D * D4 * 2;   //  8 MB
  float*  enc  = (float*)ws;  ws += (size_t)M * LAT * 4;  // ~0.4 MB
  float*  outb = qbuf;  // q is dead after flash attention; reuse

  // Embedding + Leech lattice round trip
  embed_kernel<<<M, 256, 0, stream>>>(input_ids, tok_table, emb_scales,
                                      emb_norm, pos_emb, x, T, D);
  leech_encode_kernel<<<M, 256, 0, stream>>>(x, W_el, b_el, ecs, epres, enc, D);
  leech_decode_kernel<<<M, 256, 0, stream>>>(x, enc, W_le, b_le, ecs, epres, D);

  const dim3 gridD(D / 64, M / 64);
  const dim3 gridD4(D4 / 64, M / 64);
  const int convDxD  = (D * D + 255) / 256;
  const int convDxD4 = (D * D4 + 255) / 256;
  const size_t nMD = (size_t)M * D;
  const int eltMD = (int)((nMD + 255) / 256);

  for (int i = 0; i < L; ++i) {
    // --- spectral attention ---
    layernorm_kernel<<<M, 256, 0, stream>>>(x, ln1_w + i * D, ln1_b + i * D, h,
                                            hb, D);
    convert_transpose_kernel<<<convDxD, 256, 0, stream>>>(qW + (size_t)i * D * D, wbT, D, D);
    gemm_bf16_kernel<0><<<gridD, 256, 0, stream>>>(hb, wbT, qb + i * D, qbuf, nullptr, M, D, D, D);
    convert_transpose_kernel<<<convDxD, 256, 0, stream>>>(kW + (size_t)i * D * D, wbT, D, D);
    gemm_bf16_kernel<0><<<gridD, 256, 0, stream>>>(hb, wbT, kb + i * D, kbuf, nullptr, M, D, D, D);
    convert_transpose_kernel<<<convDxD, 256, 0, stream>>>(vW + (size_t)i * D * D, wbT, D, D);
    gemm_bf16_kernel<0><<<gridD, 256, 0, stream>>>(hb, wbT, vb + i * D, vbuf, nullptr, M, D, D, D);

    fft_filter_kernel<<<B * H * Dh, 256, 0, stream>>>(qbuf, alpha, fscale, i, B, T, H, Dh);
    fft_filter_kernel<<<B * H * Dh, 256, 0, stream>>>(kbuf, alpha, fscale, i, B, T, H, Dh);
    fft_filter_kernel<<<B * H * Dh, 256, 0, stream>>>(vbuf, alpha, fscale, i, B, T, H, Dh);

    flash_attn_kernel<<<dim3(T / 64, B * H), 256, 0, stream>>>(qbuf, kbuf, vbuf,
                                                               att, B, T, H, Dh);

    f32_to_bf16_kernel<<<eltMD, 256, 0, stream>>>(att, actb, nMD);
    convert_transpose_kernel<<<convDxD, 256, 0, stream>>>(oW + (size_t)i * D * D, wbT, D, D);
    gemm_bf16_kernel<0><<<gridD, 256, 0, stream>>>(actb, wbT, ob + i * D, outb, nullptr, M, D, D, D);
    resnorm_add_kernel<<<M, 256, 0, stream>>>(x, h, outb, enorm, i, D);

    // --- FFN ---
    layernorm_kernel<<<M, 256, 0, stream>>>(x, ln2_w + i * D, ln2_b + i * D, h, hb, D);
    convert_transpose_kernel<<<convDxD4, 256, 0, stream>>>(f1W + (size_t)i * D * D4, wbT, D, D4);
    gemm_bf16_kernel<1><<<gridD4, 256, 0, stream>>>(hb, wbT, f1b + i * D4, nullptr, actb, M, D4, D, D);
    convert_transpose_kernel<<<convDxD4, 256, 0, stream>>>(f2W + (size_t)i * D4 * D, wbT, D4, D);
    gemm_bf16_kernel<0><<<gridD, 256, 0, stream>>>(actb, wbT, f2b + i * D, outb, nullptr, M, D, D4, D4);
    add_kernel<<<eltMD, 256, 0, stream>>>(x, outb, nMD);
  }

  pool_classify_kernel<<<B, 256, 0, stream>>>(x, clsW, clsb, (float*)d_out, T, D, NC);
}
